// DRL_90074054132064
// MI455X (gfx1250) — matrix operand order repaired and checked by reading, not compiled, not run on hardware
//
#include <hip/hip_runtime.h>
#include <hip/hip_bf16.h>
#include <math.h>
#include <stdint.h>

#define BN 4096
#define MN 4096
#define DN 256
#define NCB 4

typedef __attribute__((ext_vector_type(16))) _Float16 v16h;
typedef __attribute__((ext_vector_type(8)))  _Float16 v8h;
typedef __attribute__((ext_vector_type(8)))  float    v8f;
typedef __attribute__((ext_vector_type(4)))  unsigned int u32x4;
typedef __attribute__((ext_vector_type(8)))  int      i32x8;
typedef __attribute__((ext_vector_type(4)))  int      i32x4;

// ---------------- WMMA fragment helpers (wave32, 16x16x32 f16 -> f32) ----------------
// A: 16(M) x 32(K), row-major f16 tile in LDS with leading dim `ld` (halves).
__device__ __forceinline__ v16h frag_a(const _Float16* t, int ld, int kk) {
  int lane = threadIdx.x & 31;
  int r = lane & 15, kh = lane >> 4;
  const _Float16* p = t + r * ld + kk;
  v8h lo = *(const v8h*)(p + 8 * kh);
  v8h hi = *(const v8h*)(p + 16 + 8 * kh);
  v16h a;
#pragma unroll
  for (int j = 0; j < 8; ++j) { a[j] = lo[j]; a[j + 8] = hi[j]; }
  return a;
}

// B staged as B^T: 16(N) x 32(K) row-major. lane = N, kh selects K octets (8kh..8kh+7, 16+8kh..16+8kh+7).
__device__ __forceinline__ v16h frag_b(const _Float16* t, int ld, int kk) {
  int lane = threadIdx.x & 31;
  int n = lane & 15, kh = lane >> 4;
  const _Float16* p = t + n * ld + kk;
  v8h lo = *(const v8h*)(p + 8 * kh);
  v8h hi = *(const v8h*)(p + 16 + 8 * kh);
  v16h b;
#pragma unroll
  for (int j = 0; j < 8; ++j) { b[j] = lo[j]; b[j + 8] = hi[j]; }
  return b;
}

// 16x16 tile, K=256 fully unrolled (8 wmma ops). Both operands row-major, ld=264 (pad).
__device__ __forceinline__ v8f mma_k256(const _Float16* lA, const _Float16* lB) {
  v8f c = {};
#pragma unroll
  for (int kk = 0; kk < 256; kk += 32) {
    v16h a = frag_a(lA, 264, kk);
    v16h b = frag_b(lB, 264, kk);
    c = __builtin_amdgcn_wmma_f32_16x16x32_f16(false, a, false, b, (short)0, c, false, false);
  }
  return c;
}

__device__ __forceinline__ void stage16x256(const _Float16* __restrict__ g, int gld,
                                            _Float16* __restrict__ l, int tid, int tcount) {
  for (int idx = tid; idx < 16 * 32; idx += tcount) {
    int r = idx >> 5, ch = idx & 31;
    *(v8h*)(l + r * 264 + ch * 8) = *(const v8h*)(g + (size_t)r * gld + ch * 8);
  }
}

__device__ __forceinline__ unsigned long long shflx_u64(unsigned long long v, int m) {
  int lo = __shfl_xor((int)(unsigned)v, m, 32);
  int hi = __shfl_xor((int)(unsigned)(v >> 32), m, 32);
  return ((unsigned long long)(unsigned)hi << 32) | (unsigned)lo;
}

__device__ __forceinline__ unsigned ord_f32(float f) {
  unsigned u = __float_as_uint(f);
  return (u & 0x80000000u) ? ~u : (u | 0x80000000u);
}

// ---------------- Tensor Data Mover: 2D tile (rows x 32 halves) global -> LDS ----------------
// D# per CDNA5 ISA 8.3/8.4. data_size=2B; tensor 4096x4096 halves, dim0 stride 4096;
// tile_dim0=32 halves; pad_interval=16 DWORDs (one 64B row), pad_amount=4 DWORDs (8 halves)
// -> LDS row stride = 40 halves, matching frag_a/frag_b ld=40.
// amdgpu-toolchain (clang-23) 6-arg builtin: (u32x4, i32x8, i32x4, i32x4, i32x8, i32 cpol).
__device__ __forceinline__ void tdm_load_tile(unsigned lds_off_bytes,
                                              const _Float16* gsrc, int rows) {
  unsigned long long ga = (unsigned long long)(uintptr_t)gsrc;
  u32x4 g0;
  g0[0] = 1u;                                                  // count=1, user descriptor
  g0[1] = lds_off_bytes;                                       // lds_addr (bytes)
  g0[2] = (unsigned)ga;                                        // global_addr[31:0]
  g0[3] = (unsigned)((ga >> 32) & 0x01FFFFFFu) | (2u << 30);   // global_addr[56:32], type=2
  i32x8 g1;
  g1[0] = (1 << 16) | (1 << 20) | (3 << 22) | (3 << 25);       // data_size=2B, pad_en, int=16dw, amt=4dw
  g1[1] = (int)((MN & 0xFFFF) << 16);                          // tensor_dim0 lo16 (abar addr = 0)
  g1[2] = (int)(((MN >> 16) & 0xFFFF) | ((MN & 0xFFFF) << 16));// tensor_dim0 hi16 | tensor_dim1 lo16
  g1[3] = (int)(((MN >> 16) & 0xFFFF) | (32 << 16));           // tensor_dim1 hi16 | tile_dim0=32
  g1[4] = rows & 0xFFFF;                                       // tile_dim1 = rows, tile_dim2=0
  g1[5] = MN;                                                  // tensor_dim0_stride lo32
  g1[6] = 0;
  g1[7] = 0;
  i32x4 gz;  gz[0] = 0; gz[1] = 0; gz[2] = 0; gz[3] = 0;       // groups 2/3 unused (2D tile)
  i32x8 gz8;
#pragma unroll
  for (int j = 0; j < 8; ++j) gz8[j] = 0;
  __builtin_amdgcn_tensor_load_to_lds(g0, g1, gz, gz, gz8, 0);
}

// ---------------- VQ distance + argmin (fused epilogue) ----------------
__global__ void __launch_bounds__(128)
k_dist_argmin(const _Float16* __restrict__ R16, const _Float16* __restrict__ E16,
              const float* __restrict__ enorm, unsigned long long* __restrict__ argm) {
  __shared__ _Float16 lA[16 * 264];
  __shared__ _Float16 lB[4][16 * 264];
  int tid = threadIdx.x, wave = tid >> 5, lane = tid & 31;
  int row0 = blockIdx.y * 16;
  int col0 = blockIdx.x * 64 + wave * 16;
  stage16x256(R16 + (size_t)row0 * DN, DN, lA, tid, 128);
  stage16x256(E16 + (size_t)col0 * DN, DN, lB[wave], lane, 32);
  __syncthreads();
  v8f c = mma_k256(lA, lB[wave]);
  int kh = lane >> 4, n = lane & 15;
  int col = col0 + n;
  float en = enorm[col];
#pragma unroll
  for (int r = 0; r < 8; ++r) {
    int m = row0 + r + 8 * kh;
    float key = en - 2.0f * c[r];
    unsigned long long packed = ((unsigned long long)ord_f32(key) << 32) | (unsigned)col;
#pragma unroll
    for (int s = 1; s < 16; s <<= 1) {
      unsigned long long o = shflx_u64(packed, s);
      packed = (o < packed) ? o : packed;
    }
    if (n == 0) atomicMin(&argm[m], packed);
  }
}

// ---------------- S = X @ E^T  (f32 out) ----------------
__global__ void __launch_bounds__(128)
k_gemm_xet(const _Float16* __restrict__ X16, const _Float16* __restrict__ E16,
           float* __restrict__ S) {
  __shared__ _Float16 lA[16 * 264];
  __shared__ _Float16 lB[4][16 * 264];
  int tid = threadIdx.x, wave = tid >> 5, lane = tid & 31;
  int row0 = blockIdx.y * 16;
  int col0 = blockIdx.x * 64 + wave * 16;
  stage16x256(X16 + (size_t)row0 * DN, DN, lA, tid, 128);
  stage16x256(E16 + (size_t)col0 * DN, DN, lB[wave], lane, 32);
  __syncthreads();
  v8f c = mma_k256(lA, lB[wave]);
  int kh = lane >> 4, n = lane & 15;
  int col = col0 + n;
#pragma unroll
  for (int r = 0; r < 8; ++r)
    S[(size_t)(row0 + r + 8 * kh) * MN + col] = c[r];
}

// ---------------- attn: out += emb @ Wc^T + bc ----------------
__global__ void __launch_bounds__(128)
k_attn_gemm(const _Float16* __restrict__ A16, const _Float16* __restrict__ W16,
            const float* __restrict__ bc, float* __restrict__ out) {
  __shared__ _Float16 lA[16 * 264];
  __shared__ _Float16 lB[4][16 * 264];
  int tid = threadIdx.x, wave = tid >> 5, lane = tid & 31;
  int row0 = blockIdx.y * 16;
  int col0 = blockIdx.x * 64 + wave * 16;
  stage16x256(A16 + (size_t)row0 * DN, DN, lA, tid, 128);
  stage16x256(W16 + (size_t)col0 * DN, DN, lB[wave], lane, 32);
  __syncthreads();
  v8f c = mma_k256(lA, lB[wave]);
  int kh = lane >> 4, n = lane & 15;
  int col = col0 + n;
  float bv = bc[col];
#pragma unroll
  for (int r = 0; r < 8; ++r) {
    int m = row0 + r + 8 * kh;
    out[(size_t)m * DN + col] += c[r] + bv;
  }
}

// ---------------- Scode = P @ Lq^T + Q @ Lp^T (K = 4096) ----------------
// 64x128 block tile, 8 waves in 2x4 grid, each wave 32x32 (2x2 WMMA tiles).
// Operand staging via Tensor Data Mover, double buffered; fused global-min epilogue.
__global__ void __launch_bounds__(256)
k_gemm_scode(const _Float16* __restrict__ P, const _Float16* __restrict__ Lq,
             const _Float16* __restrict__ Q, const _Float16* __restrict__ Lp,
             float* __restrict__ Scode, unsigned* __restrict__ minRed) {
  __shared__ _Float16 sP[2][64 * 40], sQ[2][64 * 40];
  __shared__ _Float16 sLq[2][128 * 40], sLp[2][128 * 40];
  int tid = threadIdx.x, wave = tid >> 5, lane = tid & 31;
  int wy = wave >> 2, wx = wave & 3;
  int row0 = blockIdx.y * 64, col0 = blockIdx.x * 128;
  v8f c00 = {}, c01 = {}, c10 = {}, c11 = {};

  if (tid == 0) {                       // TDM issues once per workgroup (EXEC ignored)
    tdm_load_tile((unsigned)(uintptr_t)&sP[0][0],  P  + (size_t)row0 * MN, 64);
    tdm_load_tile((unsigned)(uintptr_t)&sQ[0][0],  Q  + (size_t)row0 * MN, 64);
    tdm_load_tile((unsigned)(uintptr_t)&sLq[0][0], Lq + (size_t)col0 * MN, 128);
    tdm_load_tile((unsigned)(uintptr_t)&sLp[0][0], Lp + (size_t)col0 * MN, 128);
  }
  for (int k0 = 0; k0 < MN; k0 += 32) {
    int buf = (k0 >> 5) & 1;
    if (tid == 0) {
      if (k0 + 32 < MN) {               // issue next chunk, then wait for current (in-order)
        int nb = buf ^ 1;
        tdm_load_tile((unsigned)(uintptr_t)&sP[nb][0],  P  + (size_t)row0 * MN + k0 + 32, 64);
        tdm_load_tile((unsigned)(uintptr_t)&sQ[nb][0],  Q  + (size_t)row0 * MN + k0 + 32, 64);
        tdm_load_tile((unsigned)(uintptr_t)&sLq[nb][0], Lq + (size_t)col0 * MN + k0 + 32, 128);
        tdm_load_tile((unsigned)(uintptr_t)&sLp[nb][0], Lp + (size_t)col0 * MN + k0 + 32, 128);
        __builtin_amdgcn_s_wait_tensorcnt(4);
      } else {
        __builtin_amdgcn_s_wait_tensorcnt(0);
      }
    }
    __syncthreads();
    const _Float16* bP  = &sP[buf][0]  + (wy * 32) * 40;
    const _Float16* bQ  = &sQ[buf][0]  + (wy * 32) * 40;
    const _Float16* bLq = &sLq[buf][0] + (wx * 32) * 40;
    const _Float16* bLp = &sLp[buf][0] + (wx * 32) * 40;
    v16h a0 = frag_a(bP, 40, 0);
    v16h a1 = frag_a(bP + 16 * 40, 40, 0);
    v16h b0 = frag_b(bLq, 40, 0);
    v16h b1 = frag_b(bLq + 16 * 40, 40, 0);
    c00 = __builtin_amdgcn_wmma_f32_16x16x32_f16(false, a0, false, b0, (short)0, c00, false, false);
    c01 = __builtin_amdgcn_wmma_f32_16x16x32_f16(false, a0, false, b1, (short)0, c01, false, false);
    c10 = __builtin_amdgcn_wmma_f32_16x16x32_f16(false, a1, false, b0, (short)0, c10, false, false);
    c11 = __builtin_amdgcn_wmma_f32_16x16x32_f16(false, a1, false, b1, (short)0, c11, false, false);
    a0 = frag_a(bQ, 40, 0);
    a1 = frag_a(bQ + 16 * 40, 40, 0);
    b0 = frag_b(bLp, 40, 0);
    b1 = frag_b(bLp + 16 * 40, 40, 0);
    c00 = __builtin_amdgcn_wmma_f32_16x16x32_f16(false, a0, false, b0, (short)0, c00, false, false);
    c01 = __builtin_amdgcn_wmma_f32_16x16x32_f16(false, a0, false, b1, (short)0, c01, false, false);
    c10 = __builtin_amdgcn_wmma_f32_16x16x32_f16(false, a1, false, b0, (short)0, c10, false, false);
    c11 = __builtin_amdgcn_wmma_f32_16x16x32_f16(false, a1, false, b1, (short)0, c11, false, false);
    __syncthreads();
  }
  int kh = lane >> 4, n = lane & 15;
  float lm = 3.4e38f;
  v8f cs[2][2] = {{c00, c01}, {c10, c11}};
#pragma unroll
  for (int ry = 0; ry < 2; ++ry)
#pragma unroll
    for (int rx = 0; rx < 2; ++rx) {
      int colb = col0 + wx * 32 + rx * 16 + n;
#pragma unroll
      for (int r = 0; r < 8; ++r) {
        int m = row0 + wy * 32 + ry * 16 + r + 8 * kh;
        float v = cs[ry][rx][r];
        Scode[(size_t)m * MN + colb] = v;
        lm = fminf(lm, v);
      }
    }
#pragma unroll
  for (int s = 1; s < 32; s <<= 1) lm = fminf(lm, __shfl_xor(lm, s, 32));
  if (lane == 0) atomicMin(minRed, ord_f32(lm));
}

// ---------------- elementwise / reduction kernels ----------------
__global__ void k_prep_sem(const float* __restrict__ semP, const float* __restrict__ semL,
                           _Float16* __restrict__ s16P, _Float16* __restrict__ s16L,
                           float* __restrict__ xnP, float* __restrict__ xnL,
                           float* __restrict__ resP, float* __restrict__ resL,
                           _Float16* __restrict__ r16P, _Float16* __restrict__ r16L) {
  __shared__ float red[256];
  int row = blockIdx.x, t = threadIdx.x;
  size_t idx = (size_t)row * DN + t;
  float v = (blockIdx.y ? semL : semP)[idx];
  if (blockIdx.y) { s16L[idx] = (_Float16)v; resL[idx] = v; r16L[idx] = (_Float16)v; }
  else            { s16P[idx] = (_Float16)v; resP[idx] = v; r16P[idx] = (_Float16)v; }
  red[t] = v * v; __syncthreads();
  for (int s = 128; s > 0; s >>= 1) { if (t < s) red[t] += red[t + s]; __syncthreads(); }
  if (t == 0) { if (blockIdx.y) xnL[row] = red[0]; else xnP[row] = red[0]; }
}

__global__ void k_prep_emb(const float* __restrict__ E, _Float16* __restrict__ E16,
                           float* __restrict__ enorm) {
  __shared__ float red[256];
  int row = blockIdx.x, t = threadIdx.x;
  size_t idx = (size_t)row * DN + t;
  float v = E[idx];
  E16[idx] = (_Float16)v;
  red[t] = v * v; __syncthreads();
  for (int s = 128; s > 0; s >>= 1) { if (t < s) red[t] += red[t + s]; __syncthreads(); }
  if (t == 0) enorm[row] = red[0];
}

__global__ void k_vq_update(const unsigned long long* __restrict__ argm,
                            const float* __restrict__ Efp,
                            float* __restrict__ res, _Float16* __restrict__ res16,
                            float* __restrict__ q, _Float16* __restrict__ emb16) {
  int b = blockIdx.x, t = threadIdx.x;
  unsigned mi = (unsigned)(argm[b] & 0xFFFFFFFFull);
  size_t idx = (size_t)b * DN + t;
  float e = Efp[(size_t)mi * DN + t];
  float r = res[idx] - e;
  res[idx] = r; res16[idx] = (_Float16)r;
  q[idx] += e;
  emb16[idx] = (_Float16)e;
}

__global__ void k_softmax_row(const float* __restrict__ S, const float* __restrict__ xn,
                              const float* __restrict__ en,
                              _Float16* __restrict__ P16, _Float16* __restrict__ L16) {
  __shared__ float red[256];
  int a = blockIdx.x, t = threadIdx.x;
  float xv = xn[a];
  float z[16];
  float mx = -3.4e38f;
#pragma unroll
  for (int j = 0; j < 16; ++j) {
    int m = t + j * 256;
    float d = fmaxf(xv + en[m] - 2.0f * S[(size_t)a * MN + m], 0.0f);
    z[j] = -sqrtf(d);
    mx = fmaxf(mx, z[j]);
  }
  red[t] = mx; __syncthreads();
  for (int s = 128; s > 0; s >>= 1) { if (t < s) red[t] = fmaxf(red[t], red[t + s]); __syncthreads(); }
  float Mx = red[0]; __syncthreads();
  float sum = 0.0f;
#pragma unroll
  for (int j = 0; j < 16; ++j) sum += expf(z[j] - Mx);
  red[t] = sum; __syncthreads();
  for (int s = 128; s > 0; s >>= 1) { if (t < s) red[t] += red[t + s]; __syncthreads(); }
  float inv = 1.0f / red[0];
#pragma unroll
  for (int j = 0; j < 16; ++j) {
    int m = t + j * 256;
    float p = expf(z[j] - Mx) * inv;
    P16[(size_t)a * MN + m] = (_Float16)p;
    L16[(size_t)a * MN + m] = (_Float16)logf(p + 1e-10f);
  }
}

__global__ void k_scode_rowloss(const float* __restrict__ Scode,
                                const unsigned* __restrict__ minRed,
                                float* __restrict__ lcmcm) {
  __shared__ float red[256];
  int a = blockIdx.x, t = threadIdx.x;
  unsigned o = minRed[0];
  unsigned ub = (o & 0x80000000u) ? (o & 0x7FFFFFFFu) : ~o;
  float Mx = -__uint_as_float(ub);                 // MaxScode = -min(Scode)
  float s = 0.0f;
#pragma unroll
  for (int j = 0; j < 16; ++j) s += expf(Scode[(size_t)a * MN + t + j * 256] + Mx);
  red[t] = s; __syncthreads();
  for (int st = 128; st > 0; st >>= 1) { if (t < st) red[t] += red[t + st]; __syncthreads(); }
  if (t == 0) {
    float denom = red[0] + 1e-5f;
    atomicAdd(lcmcm, -(Scode[(size_t)a * MN + a] + Mx - logf(denom)) / (float)BN);
  }
}

// Wc = W_out @ W_v ; bc = W_out @ b_v + b_out   (q/k proj dead: softmax over single key == 1)
__global__ void k_combine_w(const float* __restrict__ w_in, const float* __restrict__ b_in,
                            const float* __restrict__ w_out, const float* __restrict__ b_out,
                            _Float16* __restrict__ Wc16, float* __restrict__ bc) {
  __shared__ float red[256];
  int n = blockIdx.x, t = threadIdx.x;
  const float* wv = w_in + 2 * DN * DN;
  const float* bv = b_in + 2 * DN;
  float acc = 0.0f;
  for (int j = 0; j < DN; ++j) acc += w_out[(size_t)n * DN + j] * wv[(size_t)j * DN + t];
  Wc16[(size_t)n * DN + t] = (_Float16)acc;
  red[t] = w_out[(size_t)n * DN + t] * bv[t]; __syncthreads();
  for (int s = 128; s > 0; s >>= 1) { if (t < s) red[t] += red[t + s]; __syncthreads(); }
  if (t == 0) bc[n] = red[0] + b_out[n];
}

__global__ void k_final(const float* __restrict__ qP, const float* __restrict__ qL,
                        const float* __restrict__ aP, const float* __restrict__ aL,
                        const float* __restrict__ lc, float* __restrict__ out) {
  size_t i = (size_t)blockIdx.x * 256 + threadIdx.x;
  out[i] = qP[i] + qL[i] + aP[i] + aL[i] + lc[0];
}

// ---------------- host orchestration ----------------
extern "C" void kernel_launch(void* const* d_in, const int* in_sizes, int n_in,
                              void* d_out, int out_size, void* d_ws, size_t ws_size,
                              hipStream_t stream) {
  (void)in_sizes; (void)n_in; (void)out_size; (void)ws_size;
  const float* pcf = (const float*)d_in[0];
  const float* plm = (const float*)d_in[1];
  const float* emb = (const float*)d_in[2];
  const float* pwi = (const float*)d_in[3];
  const float* pbi = (const float*)d_in[4];
  const float* pwo = (const float*)d_in[5];
  const float* pbo = (const float*)d_in[6];
  const float* lwi = (const float*)d_in[7];
  const float* lbi = (const float*)d_in[8];
  const float* lwo = (const float*)d_in[9];
  const float* lbo = (const float*)d_in[10];

  char* w = (char*)d_ws;
  size_t off = 0;
  auto alloc = [&](size_t bytes) -> void* {
    void* p = w + off; off += (bytes + 255) & ~(size_t)255; return p;
  };
  const size_t BD = (size_t)BN * DN;
  const size_t MD = (size_t)MN * DN;
  const size_t BM = (size_t)BN * MN;

  _Float16* s16P = (_Float16*)alloc(BD * 2);
  _Float16* s16L = (_Float16*)alloc(BD * 2);
  _Float16* E16  = (_Float16*)alloc((size_t)NCB * MD * 2);
  float* enorm = (float*)alloc((size_t)NCB * MN * 4);
  float* xnP = (float*)alloc(BN * 4);
  float* xnL = (float*)alloc(BN * 4);
  float* resP = (float*)alloc(BD * 4);
  float* resL = (float*)alloc(BD * 4);
  _Float16* r16P = (_Float16*)alloc(BD * 2);
  _Float16* r16L = (_Float16*)alloc(BD * 2);
  float* qP = (float*)alloc(BD * 4);
  float* qL = (float*)alloc(BD * 4);
  _Float16* embP16 = (_Float16*)alloc((size_t)NCB * BD * 2);
  _Float16* embL16 = (_Float16*)alloc((size_t)NCB * BD * 2);
  unsigned long long* argm = (unsigned long long*)alloc(BN * 8);
  float* Sbuf = (float*)alloc(BM * 4);            // reused as Scode ([B,M] == [B,B])
  _Float16* P16  = (_Float16*)alloc(BM * 2);
  _Float16* Lp16 = (_Float16*)alloc(BM * 2);
  _Float16* Q16  = (_Float16*)alloc(BM * 2);
  _Float16* Lq16 = (_Float16*)alloc(BM * 2);
  unsigned* minRed = (unsigned*)alloc(256);
  float* lcmcm = (float*)alloc(256);
  _Float16* Wc16 = (_Float16*)alloc((size_t)DN * DN * 2);
  float* bc = (float*)alloc(DN * 4);
  float* attnP = (float*)alloc(BD * 4);
  float* attnL = (float*)alloc(BD * 4);

  (void)hipMemsetAsync(qP, 0, BD * 4, stream);
  (void)hipMemsetAsync(qL, 0, BD * 4, stream);
  (void)hipMemsetAsync(attnP, 0, BD * 4, stream);
  (void)hipMemsetAsync(attnL, 0, BD * 4, stream);
  (void)hipMemsetAsync(lcmcm, 0, 4, stream);

  k_prep_sem<<<dim3(BN, 2), 256, 0, stream>>>(pcf, plm, s16P, s16L, xnP, xnL,
                                              resP, resL, r16P, r16L);
  k_prep_emb<<<NCB * MN, 256, 0, stream>>>(emb, E16, enorm);

  // residual VQ (sequential over codebooks; pcf then plm)
  for (int mod = 0; mod < 2; ++mod) {
    float* res = mod ? resL : resP;
    _Float16* r16 = mod ? r16L : r16P;
    float* q = mod ? qL : qP;
    _Float16* e16 = mod ? embL16 : embP16;
    for (int i = 0; i < NCB; ++i) {
      (void)hipMemsetAsync(argm, 0xFF, BN * 8, stream);
      k_dist_argmin<<<dim3(MN / 64, BN / 16), 128, 0, stream>>>(
          r16, E16 + (size_t)i * MD, enorm + (size_t)i * MN, argm);
      k_vq_update<<<BN, 256, 0, stream>>>(argm, emb + (size_t)i * MD, res, r16, q,
                                          e16 + (size_t)i * BD);
    }
  }

  // CMCM loss per codebook
  for (int i = 0; i < NCB; ++i) {
    (void)hipMemsetAsync(minRed, 0xFF, 4, stream);
    k_gemm_xet<<<dim3(MN / 64, BN / 16), 128, 0, stream>>>(s16P, E16 + (size_t)i * MD, Sbuf);
    k_softmax_row<<<BN, 256, 0, stream>>>(Sbuf, xnP, enorm + (size_t)i * MN, P16, Lp16);
    k_gemm_xet<<<dim3(MN / 64, BN / 16), 128, 0, stream>>>(s16L, E16 + (size_t)i * MD, Sbuf);
    k_softmax_row<<<BN, 256, 0, stream>>>(Sbuf, xnL, enorm + (size_t)i * MN, Q16, Lq16);
    k_gemm_scode<<<dim3(BN / 128, BN / 64), 256, 0, stream>>>(P16, Lq16, Q16, Lp16, Sbuf, minRed);
    k_scode_rowloss<<<BN, 256, 0, stream>>>(Sbuf, minRed, lcmcm);
  }

  // folded attention: pcf_attn uses plm embeddings with pcf weights, and vice versa
  for (int i = 0; i < NCB; ++i) {
    k_combine_w<<<DN, 256, 0, stream>>>(pwi + (size_t)i * 3 * DN * DN, pbi + (size_t)i * 3 * DN,
                                        pwo + (size_t)i * DN * DN, pbo + (size_t)i * DN, Wc16, bc);
    k_attn_gemm<<<dim3(DN / 64, BN / 16), 128, 0, stream>>>(embL16 + (size_t)i * BD, Wc16, bc, attnP);
    k_combine_w<<<DN, 256, 0, stream>>>(lwi + (size_t)i * 3 * DN * DN, lbi + (size_t)i * 3 * DN,
                                        lwo + (size_t)i * DN * DN, lbo + (size_t)i * DN, Wc16, bc);
    k_attn_gemm<<<dim3(DN / 64, BN / 16), 128, 0, stream>>>(embP16 + (size_t)i * BD, Wc16, bc, attnL);
  }

  k_final<<<BN, 256, 0, stream>>>(qP, qL, attnP, attnL, lcmcm, (float*)d_out);
}
